// BiologicalSNNLayer_51290499449086
// MI455X (gfx1250) — compile-verified, matrix operand order checked by reading, not run on hardware
//
#include <hip/hip_runtime.h>

typedef __attribute__((ext_vector_type(16))) __bf16 v16bf;
typedef __attribute__((ext_vector_type(8)))  __bf16 v8bf;
typedef __attribute__((ext_vector_type(8)))  float  v8f;

#define B_DIM 16
#define S_DIM 4096
#define I_DIM 256
#define H_DIM 256
#define M_TOTAL (B_DIM * S_DIM)      // 65536 rows

// ---- constants from the reference ----
#define DT_C      0.1f
#define V_REST_C  (-65.0f)
#define V_THRESH  (-50.0f)
#define V_RESET_C (-65.0f)
#define TAU_M_C   20.0f
#define TAU_AD    100.0f
#define AD_A      0.5f
#define AD_B      0.1f
#define E_NA_C    50.0f
#define E_K_C     (-77.0f)
#define E_L_C     (-54.4f)
#define M0_C      0.05f
#define H0_C      0.6f
#define N0_C      0.32f

// ----------------------------------------------------------------------------
// Kernel 1: convert W (H x I, f32) to bf16, packed in WMMA B-fragment order.
// Layout: block (ntile 0..15, kstep 0..7) -> 1KB = 32 lanes x 32B.
// Lane's 16 bf16 (ISA 7.12.2 16-bit B layout): row = ntile*16 + lane%16,
// k = kstep*32 + (lane/16)*8 + {0..7} and + 16 + {0..7}.
// grid(128) x block(32): one wave packs one 1KB block.
// ----------------------------------------------------------------------------
__global__ __launch_bounds__(32) void w_pack_bf16(const float* __restrict__ W,
                                                  __bf16* __restrict__ Wp) {
    const int blk  = blockIdx.x;            // (ntile<<3) | kstep
    const int lane = threadIdx.x;           // 0..31
    const int row  = ((blk >> 3) << 4) + (lane & 15);
    const int kb   = ((blk & 7) << 5) + ((lane >> 4) << 3);
    const float* src = W + (size_t)row * I_DIM + kb;

    v8bf lo, hi;
    #pragma unroll
    for (int e = 0; e < 8; ++e) lo[e] = (__bf16)src[e];
    #pragma unroll
    for (int e = 0; e < 8; ++e) hi[e] = (__bf16)src[16 + e];

    __bf16* dst = Wp + (size_t)blk * 512 + lane * 16;   // lane-contiguous 32B
    *(v8bf*)(dst)     = lo;
    *(v8bf*)(dst + 8) = hi;
}

// ----------------------------------------------------------------------------
// Kernel 2: bf16 WMMA GEMM (psp = x * W^T) fused with the HH/LIF epilogue.
// Block = 256 threads = 8 waves (4 M x 2 N). Workgroup tile: 64 rows x 128 cols.
// Packed W slice (64KB) staged to LDS via global_load_async_to_lds_b128;
// B fragments fed from LDS with lane-contiguous ds_load_b128 pairs.
// ----------------------------------------------------------------------------
__global__ __launch_bounds__(256) void snn_wmma(const float* __restrict__ x,
                                                const __bf16* __restrict__ Wp,
                                                const float* __restrict__ gNa_p,
                                                const float* __restrict__ gK_p,
                                                const float* __restrict__ gL_p,
                                                float* __restrict__ spikes,
                                                float* __restrict__ volts,
                                                float* __restrict__ wout) {
    __shared__ __align__(32) __bf16 bsm[32768];   // 64 KB packed B slice

    const int lane = threadIdx.x & 31;
    const int wave = threadIdx.x >> 5;          // 0..7
    const int wm   = wave & 3;                  // wave row   0..3
    const int wn   = wave >> 2;                 // wave col   0..1

    const int m0 = blockIdx.x * 64 + wm * 16;   // this wave's row base
    const int n0 = blockIdx.y * 128 + wn * 64;  // this wave's col base

    // ---- async stage: 64KB contiguous packed-W slice -> LDS ----
    {
        const unsigned long long gbase =
            (unsigned long long)(size_t)(Wp + (size_t)blockIdx.y * 32768);
        const unsigned lds0 = (unsigned)(size_t)(&bsm[0]);
        #pragma unroll
        for (int j = 0; j < 16; ++j) {
            const unsigned boff = ((unsigned)(wave * 16 + j) << 9) + lane * 16;
            const unsigned           ldsa = lds0 + boff;
            const unsigned long long ga   = gbase + boff;
            asm volatile("global_load_async_to_lds_b128 %0, %1, off"
                         :: "v"(ldsa), "v"(ga) : "memory");
        }
        asm volatile("s_wait_asynccnt 0x0" ::: "memory");
    }
    __syncthreads();

    // 16-bit A fragment layout (ISA 7.12.2): lane holds row (lane&15),
    // K offsets kb..kb+7 and kb+16..kb+23 where kb = (lane>>4)*8.
    const int row16 = lane & 15;
    const int kb    = (lane >> 4) * 8;

    const float* xrow = x + (size_t)(m0 + row16) * I_DIM;
    const char*  bsmb = (const char*)&bsm[0];

    v8f acc0 = 0.0f, acc1 = 0.0f, acc2 = 0.0f, acc3 = 0.0f;

    #pragma unroll 2
    for (int k = 0; k < I_DIM; k += 32) {
        const int ks = k >> 5;
        // ---- A fragment: rows of x, f32 -> bf16 on the fly ----
        v16bf a;
        {
            const float* p0 = xrow + k + kb;        // 8 consecutive f32 (32B aligned)
            const float* p1 = p0 + 16;              // next 8 consecutive
            #pragma unroll
            for (int e = 0; e < 8; ++e) a[e]     = (__bf16)p0[e];
            #pragma unroll
            for (int e = 0; e < 8; ++e) a[8 + e] = (__bf16)p1[e];
        }
        // ---- 4 B fragments from LDS (lane-contiguous 32B, conflict-free) ----
        #pragma unroll
        for (int t = 0; t < 4; ++t) {
            const int fblk = ((wn * 4 + t) << 3) + ks;          // local frag block
            v16bf b = *(const v16bf*)(bsmb + fblk * 1024 + lane * 32);
            v8f c = (t == 0) ? acc0 : (t == 1) ? acc1 : (t == 2) ? acc2 : acc3;
            c = __builtin_amdgcn_wmma_f32_16x16x32_bf16(
                    /*neg_a=*/false, a, /*neg_b=*/false, b,
                    /*c_mod=*/(short)0, c, /*reuse_a=*/false, /*reuse_b=*/false);
            if (t == 0) acc0 = c; else if (t == 1) acc1 = c;
            else if (t == 2) acc2 = c; else acc3 = c;
        }
    }

    // ---- elementwise HH + adaptive-LIF epilogue ----
    const float gNa = *gNa_p;
    const float gK  = *gK_p;
    const float gL  = *gL_p;

    const int mhalf = (lane >> 4) << 3;     // C/D layout: lanes 16..31 -> M += 8

    #pragma unroll
    for (int t = 0; t < 4; ++t) {
        v8f acc = (t == 0) ? acc0 : (t == 1) ? acc1 : (t == 2) ? acc2 : acc3;
        const int ncol = n0 + t * 16 + row16;
        #pragma unroll
        for (int r = 0; r < 8; ++r) {
            const float psp = acc[r];
            const int   m   = m0 + r + mhalf;
            const size_t idx = (size_t)m * H_DIM + ncol;

            const float V = psp;
            // HH rate constants
            const float am = 0.1f * (V + 40.0f) / (1.0f - __expf(-(V + 40.0f) * 0.1f));
            const float bm = 4.0f * __expf(-(V + 65.0f) * (1.0f / 18.0f));
            const float ah = 0.07f * __expf(-(V + 65.0f) * 0.05f);
            const float bh = 1.0f / (1.0f + __expf(-(V + 35.0f) * 0.1f));
            const float an = 0.01f * (V + 55.0f) / (1.0f - __expf(-(V + 55.0f) * 0.1f));
            const float bn = 0.125f * __expf(-(V + 65.0f) * (1.0f / 80.0f));
            // one Euler step from fixed init gates
            const float mg = M0_C + DT_C * (am * (1.0f - M0_C) - bm * M0_C);
            const float hg = H0_C + DT_C * (ah * (1.0f - H0_C) - bh * H0_C);
            const float ng = N0_C + DT_C * (an * (1.0f - N0_C) - bn * N0_C);

            const float n2 = ng * ng;
            const float I_ion = gNa * mg * mg * mg * hg * (V - E_NA_C)
                              + gK * n2 * n2 * (V - E_K_C)
                              + gL * (V - E_L_C);
            const float I_in = I_ion + psp;

            const float v  = V_REST_C + I_in * (DT_C / TAU_M_C);
            const float sp = (v >= V_THRESH) ? 1.0f : 0.0f;
            const float w  = (AD_A * (v - V_REST_C) + AD_B * sp) * (DT_C / TAU_AD);
            const float vo = (sp > 0.5f) ? V_RESET_C : v;

            spikes[idx] = sp;
            volts[idx]  = vo;
            wout[idx]   = w;
        }
    }
}

extern "C" void kernel_launch(void* const* d_in, const int* in_sizes, int n_in,
                              void* d_out, int out_size, void* d_ws, size_t ws_size,
                              hipStream_t stream) {
    const float* x   = (const float*)d_in[0];   // (16, 4096, 256) f32
    const float* W   = (const float*)d_in[1];   // (256, 256) f32
    const float* gNa = (const float*)d_in[2];
    const float* gK  = (const float*)d_in[3];
    const float* gL  = (const float*)d_in[4];

    __bf16* Wp = (__bf16*)d_ws;                 // 65536 bf16 = 128 KB scratch (packed)

    float* spikes = (float*)d_out;
    float* volts  = spikes + (size_t)M_TOTAL * H_DIM;
    float* wout   = volts  + (size_t)M_TOTAL * H_DIM;

    hipLaunchKernelGGL(w_pack_bf16, dim3(128), dim3(32), 0, stream, W, Wp);

    dim3 grid(M_TOTAL / 64, H_DIM / 128);       // (1024, 2)
    hipLaunchKernelGGL(snn_wmma, grid, dim3(256), 0, stream,
                       x, Wp, gNa, gK, gL, spikes, volts, wout);
}
